// RWNN_7842610283033
// MI455X (gfx1250) — matrix operand order repaired and checked by reading, not compiled, not run on hardware
//
#include <hip/hip_runtime.h>
#include <math.h>

// ---------------------------------------------------------------------------
// Random-wired NN forward as per-level dense GEMMs on CDNA5 WMMA.
// fp32-equivalent precision via bf16 hi/lo planes (split once per element).
// GEMM inner loop: double-buffered LDS staging via async global->LDS copies
// (ASYNCcnt) when available, software-pipelined A fragments, one barrier and
// 24 v_wmma per K-chunk.
// Workspace ~= 95.7 MB; every buffer is L2-resident (192 MB L2 on MI455X).
// ---------------------------------------------------------------------------

typedef __attribute__((ext_vector_type(16))) __bf16        v16bf;
typedef __attribute__((ext_vector_type(8)))  float         v8f;
typedef __attribute__((ext_vector_type(4)))  int           v4i;

#define RS    6400   // act plane row stride in bf16 elements
#define LSTR  40     // LDS row stride in bf16 elements (32 + 8 pad)

#if defined(__gfx1250__) && __has_builtin(__builtin_amdgcn_global_load_async_to_lds_b128) && __has_builtin(__builtin_amdgcn_s_wait_asynccnt)
#define USE_ASYNC 1
#else
#define USE_ASYNC 0
#endif

// split f32 -> (hi bf16 = truncation, lo bf16 = bf16(residual)); hi+lo ~ f32
__device__ __forceinline__ void bsplit(float x, unsigned &hi16, unsigned &lo16) {
  unsigned u = __float_as_uint(x);
  hi16 = u >> 16;
  float r = x - __uint_as_float(u & 0xffff0000u);
  lo16 = __float_as_uint(r) >> 16;
}

// one 16B global->LDS transfer (async if the toolchain exposes it)
__device__ __forceinline__ void stage_one(const unsigned short* g, unsigned short* l) {
#if USE_ASYNC
  __builtin_amdgcn_global_load_async_to_lds_b128(
      (__attribute__((address_space(1))) v4i*)g,
      (__attribute__((address_space(3))) v4i*)l, 0, 0);
#else
  *(uint4*)l = *(const uint4*)g;
#endif
}

__global__ void pack_inputs(const float* __restrict__ x,
                            unsigned short* __restrict__ ah,
                            unsigned short* __restrict__ al) {
  int b = blockIdx.x;
  int c = threadIdx.x;
  if (c < 128) {
    unsigned h, l;
    bsplit(x[b * 128 + c], h, l);
    ah[(size_t)b * RS + c] = (unsigned short)h;
    al[(size_t)b * RS + c] = (unsigned short)l;
  } else if (c == 128) {                       // bias node
    ah[(size_t)b * RS + 128] = 0x3F80;         // bf16(1.0)
    al[(size_t)b * RS + 128] = 0;
  }
}

__global__ void zero_f32(float* __restrict__ p, int n) {
  int i = blockIdx.x * blockDim.x + threadIdx.x;
  if (i < n) p[i] = 0.0f;
}

// Scatter CSR edge weights into dense fp32 W (duplicate parents occur ->
// must accumulate). Column remap: inputs 0..128, later levels shifted +31
// so every level start is 32-aligned in the act planes.
__global__ void scatter_w(const int* __restrict__ po, const int* __restrict__ pids,
                          const float* __restrict__ w, float* __restrict__ Wd,
                          int levelStart, int nEdges, int fan, int Kpad) {
  int e = blockIdx.x * blockDim.x + threadIdx.x;
  if (e >= nEdges) return;
  int n  = e / fan;
  int s  = e - n * fan;
  int pe = po[levelStart + n] + s;
  int p  = pids[pe];
  int col = (p < 129) ? p : p + 31;
  atomicAdd(&Wd[(size_t)n * Kpad + col], w[pe]);
}

__global__ void convert_w(const float* __restrict__ W32,
                          unsigned short* __restrict__ Whi,
                          unsigned short* __restrict__ Wlo, int n) {
  int i = blockIdx.x * blockDim.x + threadIdx.x;
  if (i < n) {
    unsigned h, l;
    bsplit(W32[i], h, l);
    Whi[i] = (unsigned short)h;
    Wlo[i] = (unsigned short)l;
  }
}

// One level: D[node, batch] = Wdense x actT^T. WG = 8 waves; WG tile
// 128 nodes x 128 batch; per-wave 16x16 D tiles x 8. Double-buffered LDS
// B staging (async when available), pipelined A fragments, 3 WMMAs/subtile.
__global__ __launch_bounds__(256) void gemm_level(
    const unsigned short* __restrict__ Whi, const unsigned short* __restrict__ Wlo,
    const unsigned short* __restrict__ ahp, const unsigned short* __restrict__ alp,
    unsigned short* __restrict__ ahw, unsigned short* __restrict__ alw,
    float* __restrict__ out,
    int Kpad, int nodes, int colBase, int mblocks, int isOutput)
{
  __shared__ __align__(16) unsigned short lhs[2][128 * LSTR];  // hi tiles [n][k]
  __shared__ __align__(16) unsigned short lls[2][128 * LSTR];  // lo tiles [n][k]

  const int mb   = blockIdx.x % mblocks;
  const int nb   = blockIdx.x / mblocks;
  const int wave = threadIdx.x >> 5;
  const int lane = threadIdx.x & 31;
  const int hf   = lane >> 4;            // K-half select per ISA A/B layouts
  const int l15  = lane & 15;
  const int m0   = mb * 128 + wave * 16;
  const int n0   = nb * 128;

  v8f acc[8] = {};

  // A fragment base pointers (row m0+l15, K-half hf*8)
  const unsigned short* wrh = Whi + (size_t)(m0 + l15) * Kpad + hf * 8;
  const unsigned short* wrl = Wlo + (size_t)(m0 + l15) * Kpad + hf * 8;

  // staging assignment: thread moves 4 x 16B per chunk (2 rows x 2 planes)
  const int nn0  = threadIdx.x >> 2;
  const int seg8 = (threadIdx.x & 3) * 8;
  const unsigned short* sh0 = ahp + (size_t)(n0 + nn0) * RS + seg8;
  const unsigned short* sh1 = sh0 + (size_t)64 * RS;
  const unsigned short* sl0 = alp + (size_t)(n0 + nn0) * RS + seg8;
  const unsigned short* sl1 = sl0 + (size_t)64 * RS;
  const int d0 = nn0 * LSTR + seg8;
  const int d1 = (nn0 + 64) * LSTR + seg8;

  auto stage = [&](int k, int bufi) {
    stage_one(sh0 + k, &lhs[bufi][d0]);
    stage_one(sh1 + k, &lhs[bufi][d1]);
    stage_one(sl0 + k, &lls[bufi][d0]);
    stage_one(sl1 + k, &lls[bufi][d1]);
  };

  union AB { uint4 q[2]; v16bf b; };
  AB Ahc = {}, Alc = {}, Ahn = {}, Aln = {};

  const int nchunks = Kpad >> 5;

  // prologue: stage chunk 0 into buffer 0, load A fragments for chunk 0
  stage(0, 0);
  Ahc.q[0] = *(const uint4*)(wrh);
  Ahc.q[1] = *(const uint4*)(wrh + 16);
  Alc.q[0] = *(const uint4*)(wrl);
  Alc.q[1] = *(const uint4*)(wrl + 16);

  for (int i = 0; i < nchunks; ++i) {
    const int cur = i & 1;
#if USE_ASYNC
    __builtin_amdgcn_s_wait_asynccnt(0);   // my staged copies have landed
#endif
    __syncthreads();                       // everyone's copies landed; prev bufs free

    if (i + 1 < nchunks) {                 // overlap next chunk's data movement
      const int kn = (i + 1) << 5;
      stage(kn, cur ^ 1);
      Ahn.q[0] = *(const uint4*)(wrh + kn);
      Ahn.q[1] = *(const uint4*)(wrh + kn + 16);
      Aln.q[0] = *(const uint4*)(wrl + kn);
      Aln.q[1] = *(const uint4*)(wrl + kn + 16);
    }

#pragma unroll
    for (int ns = 0; ns < 8; ++ns) {
      const int ncol = ns * 16 + l15;
      union AB Bhi, Blo;
      const unsigned short* bh = &lhs[cur][ncol * LSTR + hf * 8];
      const unsigned short* bl = &lls[cur][ncol * LSTR + hf * 8];
      Bhi.q[0] = *(const uint4*)(bh);
      Bhi.q[1] = *(const uint4*)(bh + 16);
      Blo.q[0] = *(const uint4*)(bl);
      Blo.q[1] = *(const uint4*)(bl + 16);

      acc[ns] = __builtin_amdgcn_wmma_f32_16x16x32_bf16(false, Ahc.b, false, Bhi.b,
                                                        (short)0, acc[ns], false, false);
      acc[ns] = __builtin_amdgcn_wmma_f32_16x16x32_bf16(false, Ahc.b, false, Blo.b,
                                                        (short)0, acc[ns], false, false);
      acc[ns] = __builtin_amdgcn_wmma_f32_16x16x32_bf16(false, Alc.b, false, Bhi.b,
                                                        (short)0, acc[ns], false, false);
    }

    Ahc = Ahn; Alc = Aln;                  // rotate pipelined A registers
  }

  // ---- store: C/D layout -> lane holds N=(lane&15), M = v + 8*hf
  const int nodeBase = m0 + hf * 8;
#pragma unroll
  for (int ns = 0; ns < 8; ++ns) {
    const int b = n0 + ns * 16 + l15;
    if (isOutput) {
      if (nodeBase < nodes) {
        float* o = out + (size_t)b * 64 + nodeBase;
#pragma unroll
        for (int v = 0; v < 8; ++v) o[v] = acc[ns][v];
      }
    } else {
      unsigned hh[8], ll[8];
#pragma unroll
      for (int v = 0; v < 8; ++v) {
        float t = tanhf(acc[ns][v]);
        bsplit(t, hh[v], ll[v]);
      }
      uint4 H, L;
      H.x = hh[0] | (hh[1] << 16); H.y = hh[2] | (hh[3] << 16);
      H.z = hh[4] | (hh[5] << 16); H.w = hh[6] | (hh[7] << 16);
      L.x = ll[0] | (ll[1] << 16); L.y = ll[2] | (ll[3] << 16);
      L.z = ll[4] | (ll[5] << 16); L.w = ll[6] | (ll[7] << 16);
      size_t off = (size_t)b * RS + colBase + nodeBase;   // 16B aligned
      *(uint4*)(ahw + off) = H;
      *(uint4*)(alw + off) = L;
    }
  }
}

extern "C" void kernel_launch(void* const* d_in, const int* in_sizes, int n_in,
                              void* d_out, int out_size, void* d_ws, size_t ws_size,
                              hipStream_t stream)
{
  (void)in_sizes; (void)n_in; (void)out_size; (void)ws_size;

  const float* x    = (const float*)d_in[0];   // (2048, 128) f32
  const float* wts  = (const float*)d_in[1];   // 401408 f32
  const int*   po   = (const int*)d_in[2];     // parent_offsets
  const int*   pids = (const int*)d_in[3];     // parent_ids
  float*       out  = (float*)d_out;           // (2048, 64) f32

  char* ws = (char*)d_ws;
  unsigned short* act_hi = (unsigned short*)(ws);                      // 26.2 MB
  unsigned short* act_lo = (unsigned short*)(ws + 26214400);           // 26.2 MB
  float*          W32    = (float*)         (ws + 52428800);           // 21.6 MB
  unsigned short* Whi    = (unsigned short*)(ws + 74055680);           // 10.8 MB
  unsigned short* Wlo    = (unsigned short*)(ws + 84869120);           // 10.8 MB

  pack_inputs<<<2048, 160, 0, stream>>>(x, act_hi, act_lo);

  // Global node-id level starts (structural constants of the reference graph)
  static const int LS[9] = {0, 129, 1153, 2177, 3201, 4225, 5249, 6273, 6337};

  for (int l = 1; l <= 7; ++l) {
    const int start   = LS[l];
    const int nodes   = LS[l + 1] - LS[l];       // 1024 hidden / 64 output
    const int fan     = (l == 7) ? 128 : 64;
    const int Kpad    = start + 31;              // padded act column count, mult of 32
    const int mblocks = (nodes + 127) / 128;     // 8 hidden, 1 output
    const int isOut   = (l == 7) ? 1 : 0;

    const int zn = mblocks * 128 * Kpad;
    zero_f32<<<(zn + 255) / 256, 256, 0, stream>>>(W32, zn);

    const int ne = nodes * fan;
    scatter_w<<<(ne + 255) / 256, 256, 0, stream>>>(po, pids, wts, W32,
                                                    start, ne, fan, Kpad);

    convert_w<<<(zn + 255) / 256, 256, 0, stream>>>(W32, Whi, Wlo, zn);

    gemm_level<<<mblocks * 16, 256, 0, stream>>>(Whi, Wlo, act_hi, act_lo,
                                                 act_hi, act_lo, out,
                                                 Kpad, nodes, /*colBase=*/Kpad,
                                                 mblocks, isOut);
  }
}
